// TransolverPlusPlusAttention_26139170963684
// MI455X (gfx1250) — matrix operand order repaired and checked by reading, not compiled, not run on hardware
//
#include <hip/hip_runtime.h>
#include <math.h>

// ---------------- problem constants ----------------
static const int kB = 2, kN = 32768, kH = 8, kD = 32, kG = 64, kHID = 256;
#define EPSG 1.1920929e-07f

// ---------------- WMMA types / helpers ----------------
typedef __attribute__((ext_vector_type(16))) __bf16 v16bf;
typedef __attribute__((ext_vector_type(8)))  float  v8f;

__device__ __forceinline__ v8f wmma_bf16(v16bf a, v16bf b, v8f c) {
  // emits v_wmma_f32_16x16x32_bf16
  return __builtin_amdgcn_wmma_f32_16x16x32_bf16(false, a, false, b, (short)0, c,
                                                 false, false);
}

// A fragment: 16x32 bf16, rows contiguous with stride ld (elements).
// Contiguous-k layout (B uses same): lane m=L&15 holds row m; half=L>>4: elements 0..15 = K[16h..16h+15]
// (one contiguous 32B run per lane).
__device__ __forceinline__ v16bf load_a_frag(const __bf16* src, int ld) {
  int lane = threadIdx.x & 31;
  int m = lane & 15, half = lane >> 4;
  const __bf16* row = src + (size_t)m * ld;
  v16bf a;
#pragma unroll
  for (int e = 0; e < 8; ++e) a[e] = row[16 * half + e];
#pragma unroll
  for (int e = 0; e < 8; ++e) a[8 + e] = row[16 * half + 8 + e];
  return a;
}

// Fragment from LDS pre-swizzled "fragment-major" layout: [lane][16] contiguous.
// For B (32x16 KxN subtile): element (k,n) lives at lane=(k>>4)*16+(n&15),
// elem=k&15  ->  lane reads 32 contiguous bytes (2x ds_load_b128).
__device__ __forceinline__ v16bf load_frag_contig(const __bf16* base) {
  int lane = threadIdx.x & 31;
  const __bf16* p = base + lane * 16;
  v16bf b;
#pragma unroll
  for (int e = 0; e < 16; ++e) b[e] = p[e];
  return b;
}

// swizzled destination index for B subtile element (k in 0..31, n16 in 0..15)
__device__ __forceinline__ int bswz(int k, int n16) {
  return ((k >> 4) * 16 + n16) * 16 + (k & 15);
}

__device__ __forceinline__ float gelu_exact(float x) {
  return 0.5f * x * (1.0f + erff(x * 0.70710678118654752f));
}

// ---------------- K0: weight convert + zero init ----------------
__global__ void k_prep(const float* W_in, const float* W_t1, const float* W_slice,
                       const float* W_out, __bf16* Wb_in, __bf16* Wb_t1,
                       __bf16* Wb_slice, __bf16* Wb_out, float* slice_token,
                       float* slice_norm) {
  int i = blockIdx.x * 256 + threadIdx.x;
  if (i < kHID * kH * kD) { Wb_in[i] = (__bf16)W_in[i]; Wb_out[i] = (__bf16)W_out[i]; }
  if (i < kD * kG) { Wb_t1[i] = (__bf16)W_t1[i]; Wb_slice[i] = (__bf16)W_slice[i]; }
  if (i < kB * kH * kG * kD) slice_token[i] = 0.0f;
  if (i < kB * kH * kG) slice_norm[i] = 0.0f;
}

// ---------------- K1: x_mid = x @ W_in + b_in  (M=B*N, K=256, N=256) ----------------
// block 256 threads = 8 waves; block tile 128(M) x 64(N); wave tile 32x32 (2x2 WMMA)
__global__ void k_gemm_in(const float* x, const __bf16* Wb, const float* b_in,
                          __bf16* xmid) {
  __shared__ __bf16 As[128 * 32];   // row-major (A frag is contiguous per lane)
  __shared__ __bf16 Bs[32 * 64];    // fragment-swizzled: 4 subtiles x [lane][16]
  int m0 = blockIdx.x * 128;
  int n0 = blockIdx.y * 64;
  int wave = threadIdx.x >> 5;
  int wm = wave & 3, wn = wave >> 2;
  v8f acc[2][2] = {};
  for (int k0 = 0; k0 < 256; k0 += 32) {
#pragma unroll
    for (int e = 0; e < 16; ++e) {
      int idx = threadIdx.x * 16 + e;
      int r = idx >> 5, c = idx & 31;
      As[idx] = (__bf16)x[(size_t)(m0 + r) * 256 + k0 + c];
    }
    if (k0 < 224)
      __builtin_prefetch(&x[(size_t)(m0 + ((threadIdx.x * 16) >> 5)) * 256 + k0 + 32], 0, 1);
#pragma unroll
    for (int e = 0; e < 8; ++e) {
      int idx = threadIdx.x * 8 + e;
      int r = idx >> 6, c = idx & 63;           // r=k (0..31), c=n (0..63)
      Bs[(c >> 4) * 512 + bswz(r, c & 15)] = Wb[(size_t)(k0 + r) * 256 + n0 + c];
    }
    __syncthreads();
#pragma unroll
    for (int i = 0; i < 2; ++i) {
      v16bf a = load_a_frag(&As[(wm * 32 + 16 * i) * 32], 32);
#pragma unroll
      for (int j = 0; j < 2; ++j) {
        v16bf b = load_frag_contig(&Bs[(wn * 2 + j) * 512]);
        acc[i][j] = wmma_bf16(a, b, acc[i][j]);
      }
    }
    __syncthreads();
  }
  int lane = threadIdx.x & 31, nn = lane & 15, half = lane >> 4;
#pragma unroll
  for (int i = 0; i < 2; ++i)
#pragma unroll
    for (int j = 0; j < 2; ++j)
#pragma unroll
      for (int r = 0; r < 8; ++r) {
        int M = m0 + wm * 32 + 16 * i + r + 8 * half;   // global token index b*N+n
        int col = n0 + wn * 32 + 16 * j + nn;           // h*32 + d
        float v = acc[i][j][r] + b_in[col];
        int b = M >> 15, n = M & (kN - 1);
        int h = col >> 5, d = col & 31;
        xmid[(((size_t)(b * kH + h)) * kN + n) * kD + d] = (__bf16)v;
      }
}

// ---------------- K2: routing (temperature + gumbel softmax) ----------------
// one wave handles 16 tokens of one (b,h). grid = B*H*N/16/8 blocks of 256.
__global__ void k_route(const __bf16* xmid, const __bf16* Wt1, const __bf16* Wsl,
                        const float* W_t2, const float* b_t1, const float* b_t2,
                        const float* bias, const float* b_slice, const float* u,
                        __bf16* sw, __bf16* swT, float* slice_norm) {
  __shared__ __bf16 sWt1[2048];   // 32x64, fragment-swizzled (4 subtiles)
  __shared__ __bf16 sWsl[2048];
  int t = threadIdx.x;
#pragma unroll
  for (int e = 0; e < 8; ++e) {
    int idx = t * 8 + e;
    int k = idx >> 6, c = idx & 63;
    int dst = (c >> 4) * 512 + bswz(k, c & 15);
    sWt1[dst] = Wt1[idx];
    sWsl[dst] = Wsl[idx];
  }
  __syncthreads();

  int wid = blockIdx.x * 8 + (t >> 5);
  int bh = wid >> 11;            // / (N/16)
  int n0 = (wid & 2047) * 16;
  int h = bh & 7;
  int lane = t & 31, nn = lane & 15, half = lane >> 4;

  v16bf a = load_a_frag(xmid + ((size_t)bh * kN + n0) * kD, kD);

  // ---- temperature path: p[r] = sum_g gelu(t1[m][g]) * W_t2[g]
  float p[8];
#pragma unroll
  for (int r = 0; r < 8; ++r) p[r] = 0.0f;
#pragma unroll
  for (int c = 0; c < 4; ++c) {
    v16bf bfr = load_frag_contig(&sWt1[c * 512]);
    v8f t1 = {};
    t1 = wmma_bf16(a, bfr, t1);
    int g = 16 * c + nn;
    float w2 = W_t2[g], bt = b_t1[g];
#pragma unroll
    for (int r = 0; r < 8; ++r) p[r] += gelu_exact(t1[r] + bt) * w2;
  }
#pragma unroll
  for (int mask = 1; mask < 16; mask <<= 1)
#pragma unroll
    for (int r = 0; r < 8; ++r) p[r] += __shfl_xor(p[r], mask, 32);
  float tempr[8];
  float bh_bias = bias[h], bt2 = b_t2[0];
#pragma unroll
  for (int r = 0; r < 8; ++r)
    tempr[r] = fmaxf(gelu_exact(p[r] + bt2) + bh_bias, 0.01f);

  // ---- logits + gumbel, scaled by temperature
  float z[4][8];
#pragma unroll
  for (int c = 0; c < 4; ++c) {
    v16bf bfr = load_frag_contig(&sWsl[c * 512]);
    v8f lg = {};
    lg = wmma_bf16(a, bfr, lg);
    int g = 16 * c + nn;
    float bs = b_slice[g];
#pragma unroll
    for (int r = 0; r < 8; ++r) {
      int m = r + 8 * half;
      float uu = u[((size_t)bh * kN + n0 + m) * kG + g];
      float gb = -logf(-logf(uu + EPSG) + EPSG);
      z[c][r] = (lg[r] + bs + gb) / tempr[r];
    }
  }
  // ---- softmax over g=64 (4 column tiles x 16 lanes of this half)
  float mx[8], sm[8];
#pragma unroll
  for (int r = 0; r < 8; ++r) {
    float m = z[0][r];
#pragma unroll
    for (int c = 1; c < 4; ++c) m = fmaxf(m, z[c][r]);
    mx[r] = m;
  }
#pragma unroll
  for (int mask = 1; mask < 16; mask <<= 1)
#pragma unroll
    for (int r = 0; r < 8; ++r) mx[r] = fmaxf(mx[r], __shfl_xor(mx[r], mask, 32));
#pragma unroll
  for (int r = 0; r < 8; ++r) {
    float s = 0.0f;
#pragma unroll
    for (int c = 0; c < 4; ++c) { z[c][r] = __expf(z[c][r] - mx[r]); s += z[c][r]; }
    sm[r] = s;
  }
#pragma unroll
  for (int mask = 1; mask < 16; mask <<= 1)
#pragma unroll
    for (int r = 0; r < 8; ++r) sm[r] += __shfl_xor(sm[r], mask, 32);
#pragma unroll
  for (int r = 0; r < 8; ++r) sm[r] = 1.0f / sm[r];

  // ---- emit sw, swT (bf16), accumulate slice_norm
#pragma unroll
  for (int c = 0; c < 4; ++c) {
    int g = 16 * c + nn;
    float csum = 0.0f;
#pragma unroll
    for (int r = 0; r < 8; ++r) {
      int m = r + 8 * half;
      float v = z[c][r] * sm[r];
      csum += v;
      sw[((size_t)bh * kN + n0 + m) * kG + g] = (__bf16)v;
      swT[((size_t)bh * kG + g) * kN + n0 + m] = (__bf16)v;
    }
    atomicAdd(&slice_norm[bh * kG + g], csum);
  }
}

// ---------------- K3: slice_token = swT @ x_mid (split-K, WMMA, atomics) ----------------
// grid = B*H*64 blocks of 256 (8 waves). K chunk = 512; x_mid chunk staged in LDS
// in fragment-swizzled layout so all 8 waves consume it with contiguous ds loads.
__global__ void k_dispatch(const __bf16* swT, const __bf16* xmid, float* slice_token) {
  __shared__ __bf16 Xs[512 * 32];   // [16 ksteps][2 subtiles][32 lanes][16]
  int bh = blockIdx.x >> 6;
  int split = blockIdx.x & 63;
  int t = threadIdx.x;
  const __bf16* src = xmid + ((size_t)bh * kN + split * 512) * kD;
#pragma unroll
  for (int r2 = 0; r2 < 2; ++r2) {
    int k = t * 2 + r2;                 // 0..511
    const __bf16* row = src + (size_t)k * kD;
    int kstep = k >> 5, kk = k & 31;
    int lanebase = (kk >> 4) * 16, elem = kk & 15;
#pragma unroll
    for (int d = 0; d < 32; ++d) {
      int dst = ((kstep * 2 + (d >> 4)) * 32 + lanebase + (d & 15)) * 16 + elem;
      Xs[dst] = row[d];
    }
  }
  __syncthreads();

  int wave = t >> 5;
  int mt = wave & 3, nt = wave >> 2;
  const __bf16* Ab = swT + ((size_t)bh * kG + mt * 16) * kN + split * 512;
  const __bf16* Arow = Ab + (size_t)(t & 15) * kN;
  v8f acc = {};
  for (int ks = 0; ks < 16; ++ks) {
    if (ks < 14) __builtin_prefetch(Arow + (ks + 2) * 32, 0, 1);
    v16bf a = load_a_frag(Ab + ks * 32, kN);
    v16bf b = load_frag_contig(&Xs[(ks * 2 + nt) * 512]);
    acc = wmma_bf16(a, b, acc);
  }
  int lane = t & 31, nn = lane & 15, half = lane >> 4;
#pragma unroll
  for (int r = 0; r < 8; ++r) {
    int g = mt * 16 + r + 8 * half;
    int d = nt * 16 + nn;
    atomicAdd(&slice_token[((size_t)bh * kG + g) * kD + d], acc[r]);
  }
}

// ---------------- K4: normalize + qkv + 64x64 slice attention (tiny) ----------------
__global__ void k_slice_attn(const float* slice_token, const float* slice_norm,
                             const float* W_qkv, const float* b_qkv,
                             __bf16* out_slice) {
  __shared__ float st[kG * kD];
  __shared__ float q[kG * kD], kk_[kG * kD], v[kG * kD];
  __shared__ float att[kG * kG];
  int bh = blockIdx.x, t = threadIdx.x;
  for (int i = t; i < kG * kD; i += 256) {
    int g = i >> 5;
    st[i] = slice_token[(size_t)bh * kG * kD + i] / (slice_norm[bh * kG + g] + 1e-5f);
  }
  __syncthreads();
  for (int i = t; i < kG * 96; i += 256) {
    int g = i / 96, c = i % 96;
    float s = b_qkv[c];
    for (int d = 0; d < kD; ++d) s += st[g * kD + d] * W_qkv[d * 96 + c];
    if (c < 32) q[g * kD + c] = s;
    else if (c < 64) kk_[g * kD + c - 32] = s;
    else v[g * kD + c - 64] = s;
  }
  __syncthreads();
  for (int i = t; i < kG * kG; i += 256) {
    int g = i >> 6, kc = i & 63;
    float s = 0.0f;
    for (int d = 0; d < kD; ++d) s += q[g * kD + d] * kk_[kc * kD + d];
    att[i] = s * 0.17677669529663687f;  // 1/sqrt(32)
  }
  __syncthreads();
  if (t < kG) {
    float mx = -1e30f;
    for (int kc = 0; kc < kG; ++kc) mx = fmaxf(mx, att[t * kG + kc]);
    float s = 0.0f;
    for (int kc = 0; kc < kG; ++kc) { float e = __expf(att[t * kG + kc] - mx); att[t * kG + kc] = e; s += e; }
    float inv = 1.0f / s;
    for (int kc = 0; kc < kG; ++kc) att[t * kG + kc] *= inv;
  }
  __syncthreads();
  for (int i = t; i < kG * kD; i += 256) {
    int g = i >> 5, d = i & 31;
    float s = 0.0f;
    for (int kc = 0; kc < kG; ++kc) s += att[g * kG + kc] * v[kc * kD + d];
    out_slice[(size_t)bh * kG * kD + i] = (__bf16)s;
  }
}

// ---------------- K5: out_mid[b,n,h*32+d] = sw @ out_slice ----------------
// grid = B*H*(N/128) blocks of 256; out_slice staged swizzled in LDS per block.
__global__ void k_combine(const __bf16* sw, const __bf16* out_slice, __bf16* out_mid) {
  __shared__ __bf16 Bsl[4096];   // [4 ksteps][2 subtiles][32 lanes][16]; ksteps 2-3 never read
  int bh = blockIdx.x >> 8;
  int mblk = blockIdx.x & 255;
  int t = threadIdx.x;
  const __bf16* osrc = out_slice + (size_t)bh * kG * kD;
#pragma unroll
  for (int e = 0; e < 16; ++e) {
    int idx = t * 16 + e;               // k*32 + d, k in 0..127 (k >= 64 never read)
    int k = idx >> 5, d = idx & 31;
    int dst = (((k >> 5) * 2 + (d >> 4)) * 32 + ((k & 31) >> 4) * 16 + (d & 15)) * 16 + (k & 15);
    Bsl[dst] = osrc[idx];
  }
  __syncthreads();

  int wave = t >> 5;
  int n0 = mblk * 128 + wave * 16;
  const __bf16* Ab = sw + ((size_t)bh * kN + n0) * kG;
  v8f acc[2] = {};
#pragma unroll
  for (int k0 = 0; k0 < kG; k0 += 32) {
    v16bf a = load_a_frag(Ab + k0, kG);
#pragma unroll
    for (int j = 0; j < 2; ++j) {
      v16bf b = load_frag_contig(&Bsl[((k0 >> 5) * 2 + j) * 512]);
      acc[j] = wmma_bf16(a, b, acc[j]);
    }
  }
  int lane = t & 31, nn = lane & 15, half = lane >> 4;
  int b = bh >> 3, h = bh & 7;
#pragma unroll
  for (int j = 0; j < 2; ++j)
#pragma unroll
    for (int r = 0; r < 8; ++r) {
      int row = n0 + r + 8 * half;
      int d = j * 16 + nn;
      out_mid[((size_t)b * kN + row) * kHID + h * kD + d] = (__bf16)acc[j][r];
    }
}

// ---------------- K6: out = out_mid @ W_out + b_out (f32 result) ----------------
__global__ void k_gemm_out(const __bf16* A, const __bf16* Wb, const float* b_out,
                           float* out) {
  __shared__ __bf16 As[128 * 32];
  __shared__ __bf16 Bs[32 * 64];    // fragment-swizzled
  int m0 = blockIdx.x * 128;
  int n0 = blockIdx.y * 64;
  int wave = threadIdx.x >> 5;
  int wm = wave & 3, wn = wave >> 2;
  v8f acc[2][2] = {};
  for (int k0 = 0; k0 < 256; k0 += 32) {
#pragma unroll
    for (int e = 0; e < 16; ++e) {
      int idx = threadIdx.x * 16 + e;
      int r = idx >> 5, c = idx & 31;
      As[idx] = A[(size_t)(m0 + r) * 256 + k0 + c];
    }
    if (k0 < 224)
      __builtin_prefetch(&A[(size_t)(m0 + ((threadIdx.x * 16) >> 5)) * 256 + k0 + 32], 0, 1);
#pragma unroll
    for (int e = 0; e < 8; ++e) {
      int idx = threadIdx.x * 8 + e;
      int r = idx >> 6, c = idx & 63;
      Bs[(c >> 4) * 512 + bswz(r, c & 15)] = Wb[(size_t)(k0 + r) * 256 + n0 + c];
    }
    __syncthreads();
#pragma unroll
    for (int i = 0; i < 2; ++i) {
      v16bf a = load_a_frag(&As[(wm * 32 + 16 * i) * 32], 32);
#pragma unroll
      for (int j = 0; j < 2; ++j) {
        v16bf b = load_frag_contig(&Bs[(wn * 2 + j) * 512]);
        acc[i][j] = wmma_bf16(a, b, acc[i][j]);
      }
    }
    __syncthreads();
  }
  int lane = threadIdx.x & 31, nn = lane & 15, half = lane >> 4;
#pragma unroll
  for (int i = 0; i < 2; ++i)
#pragma unroll
    for (int j = 0; j < 2; ++j)
#pragma unroll
      for (int r = 0; r < 8; ++r) {
        int M = m0 + wm * 32 + 16 * i + r + 8 * half;
        int col = n0 + wn * 32 + 16 * j + nn;
        out[(size_t)M * 256 + col] = acc[i][j][r] + b_out[col];
      }
}

// ---------------- host launcher ----------------
extern "C" void kernel_launch(void* const* d_in, const int* in_sizes, int n_in,
                              void* d_out, int out_size, void* d_ws, size_t ws_size,
                              hipStream_t stream) {
  const float* x       = (const float*)d_in[0];
  const float* u       = (const float*)d_in[1];
  const float* W_in    = (const float*)d_in[2];
  const float* b_in    = (const float*)d_in[3];
  const float* W_t1    = (const float*)d_in[4];
  const float* b_t1    = (const float*)d_in[5];
  const float* W_t2    = (const float*)d_in[6];
  const float* b_t2    = (const float*)d_in[7];
  const float* bias    = (const float*)d_in[8];
  const float* W_slice = (const float*)d_in[9];
  const float* b_slice = (const float*)d_in[10];
  const float* W_qkv   = (const float*)d_in[11];
  const float* b_qkv   = (const float*)d_in[12];
  const float* W_out   = (const float*)d_in[13];
  const float* b_out   = (const float*)d_in[14];
  float* out = (float*)d_out;

  char* ws = (char*)d_ws;
  size_t off = 0;
  auto take = [&](size_t bytes) {
    char* p = ws + off;
    off = (off + bytes + 255) & ~(size_t)255;
    return p;
  };
  const size_t tokens = (size_t)kB * kN;                        // 65536
  __bf16* xmid     = (__bf16*)take(tokens * kH * kD * 2);       // (b,h,n,d): 32MB
  __bf16* sw       = (__bf16*)take(tokens * kH * kG * 2);       // (b,h,n,g): 64MB
  __bf16* swT      = (__bf16*)take(tokens * kH * kG * 2);       // (b,h,g,n): 64MB
  __bf16* out_mid  = (__bf16*)take(tokens * kHID * 2);          // (b,n,256): 32MB
  float*  slice_tok = (float*)take((size_t)kB * kH * kG * kD * 4);
  float*  slice_nrm = (float*)take((size_t)kB * kH * kG * 4);
  __bf16* out_slice = (__bf16*)take((size_t)kB * kH * kG * kD * 2);
  __bf16* Wb_in    = (__bf16*)take((size_t)kHID * kH * kD * 2);
  __bf16* Wb_t1    = (__bf16*)take((size_t)kD * kG * 2);
  __bf16* Wb_slice = (__bf16*)take((size_t)kD * kG * 2);
  __bf16* Wb_out   = (__bf16*)take((size_t)kH * kD * kHID * 2);
  (void)ws_size; (void)in_sizes; (void)n_in; (void)out_size;

  k_prep<<<256, 256, 0, stream>>>(W_in, W_t1, W_slice, W_out, Wb_in, Wb_t1,
                                  Wb_slice, Wb_out, slice_tok, slice_nrm);
  k_gemm_in<<<dim3(512, 4), 256, 0, stream>>>(x, Wb_in, b_in, xmid);
  k_route<<<4096, 256, 0, stream>>>(xmid, Wb_t1, Wb_slice, W_t2, b_t1, b_t2,
                                    bias, b_slice, u, sw, swT, slice_nrm);
  k_dispatch<<<kB * kH * 64, 256, 0, stream>>>(swT, xmid, slice_tok);
  k_slice_attn<<<kB * kH, 256, 0, stream>>>(slice_tok, slice_nrm, W_qkv, b_qkv,
                                            out_slice);
  k_combine<<<kB * kH * (kN / 128), 256, 0, stream>>>(sw, out_slice, out_mid);
  k_gemm_out<<<dim3(512, 4), 256, 0, stream>>>(out_mid, Wb_out, b_out, out);
}
